// Encoder_20263655702890
// MI455X (gfx1250) — compile-verified, matrix operand order checked by reading, not run on hardware
//
#include <hip/hip_runtime.h>
#include <hip/hip_bf16.h>

typedef __attribute__((ext_vector_type(16))) __bf16 v16bf;
typedef __attribute__((ext_vector_type(8)))  float  v8f;
typedef __attribute__((ext_vector_type(4)))  unsigned int u32x4;
typedef __attribute__((ext_vector_type(8)))  int i32x8;
typedef __attribute__((ext_vector_type(4)))  int i32x4;

#define B_  64
#define T_  512
#define E_  256
#define H_  512
#define H3  1536
#define H6  3072
#define BT  (B_ * T_)                       // 32768
#define BTH ((size_t)B_ * T_ * H_)          // 16777216

// ---------------------------------------------------------------------------
// helpers
// ---------------------------------------------------------------------------
__device__ __forceinline__ v8f v8f_zero() {
    v8f z;
#pragma unroll
    for (int i = 0; i < 8; ++i) z[i] = 0.0f;
    return z;
}

// Load a 16-element bf16 WMMA fragment from a row-major [16][K] tile.
// p points at element [row][kbase + half*8]; elements 0..7 -> K+0..7,
// elements 8..15 -> K+16..23 (per CDNA5 16-bit A/B VGPR layout).
__device__ __forceinline__ v16bf ld_frag16(const __bf16* p) {
    v16bf r;
#pragma unroll
    for (int i = 0; i < 8; ++i) r[i] = p[i];
#pragma unroll
    for (int i = 0; i < 8; ++i) r[i + 8] = p[16 + i];
    return r;
}

__device__ __forceinline__ float sigmoidf_(float x) {
    return 1.0f / (1.0f + __expf(-x));
}
__device__ __forceinline__ float tanhf_(float x) {
    return 1.0f - 2.0f / (__expf(2.0f * x) + 1.0f);
}

// ---------------------------------------------------------------------------
// Kernel 1: weight precision/layout prep.
//  Wcat[k][n]  (n<1536: W_f, else W_b)           bf16, [256][3072]
//  Ut[d][n][k] = U_d[k][n]                       bf16, [2][1536][512]  (N-major)
// ---------------------------------------------------------------------------
__global__ void prep_kernel(const float* __restrict__ Wf, const float* __restrict__ Wb,
                            const float* __restrict__ Uf, const float* __restrict__ Ub,
                            __bf16* __restrict__ Wcat, __bf16* __restrict__ Ut) {
    const size_t total1 = (size_t)E_ * H6;          // 786432
    const size_t total2 = (size_t)2 * H3 * H_;      // 1572864
    for (size_t i = (size_t)blockIdx.x * blockDim.x + threadIdx.x;
         i < total1 + total2; i += (size_t)gridDim.x * blockDim.x) {
        if (i < total1) {
            int k = (int)(i / H6), n = (int)(i % H6);
            float v = (n < H3) ? Wf[(size_t)k * H3 + n] : Wb[(size_t)k * H3 + (n - H3)];
            Wcat[i] = (__bf16)v;
        } else {
            size_t r = i - total1;
            int d   = (int)(r / ((size_t)H3 * H_));
            size_t rem = r % ((size_t)H3 * H_);
            int n = (int)(rem / H_), k = (int)(rem % H_);
            const float* U = d ? Ub : Uf;
            Ut[r] = (__bf16)U[(size_t)k * H3 + n];
        }
    }
}

// ---------------------------------------------------------------------------
// Kernel 2: embedding gather + input projection GEMM (bf16 WMMA, f32 acc).
//   xp[t*64 + b][n] = emb[tokens[b][t]] . Wcat[:, n] + bias0[n]
//   M = 32768 (row = t*64+b), K = 256, N = 3072.
//   Block tile 128x128, 8 waves, wave = 64Mx32N (4x2 WMMA tiles), K-chunks of 32.
// ---------------------------------------------------------------------------
__global__ __launch_bounds__(256) void gemm1_kernel(
    const float* __restrict__ emb, const __bf16* __restrict__ Wcat,
    const float* __restrict__ bf0, const float* __restrict__ bb0,
    const int* __restrict__ tokens, float* __restrict__ xp) {
    __shared__ __bf16 Als[128 * 264];   // [m][k], padded row stride 264
    __shared__ __bf16 Bls[128 * 264];   // [n][k] (transposed W tile)

    const int tid = threadIdx.x;
    const int bx = blockIdx.x;          // N tile: 0..23
    const int by = blockIdx.y;          // M tile: 0..255

    // ---- stage A: gather 128 embedding rows x 256 cols, convert to bf16 ----
#pragma unroll 1
    for (int it = 0; it < 32; ++it) {
        int i   = tid + it * 256;       // float4 index, 128 rows * 64 f4/row
        int row = i >> 6;
        int c4  = (i & 63) << 2;
        int g   = by * 128 + row;       // global row = t*64 + b
        int tok = tokens[(g & 63) * T_ + (g >> 6)];
        float4 v = *(const float4*)(emb + (size_t)tok * E_ + c4);
        __bf16* d = &Als[row * 264 + c4];
        d[0] = (__bf16)v.x; d[1] = (__bf16)v.y; d[2] = (__bf16)v.z; d[3] = (__bf16)v.w;
    }
    // ---- stage B: Wcat[k][3072] -> Bls[n][k] (transpose into LDS) ----
#pragma unroll 1
    for (int it = 0; it < 64; ++it) {
        int i = tid + it * 256;         // 256 k-rows * 64 pairs
        int k = i >> 6;
        int u = i & 63;
        const __bf16* src = Wcat + (size_t)k * H6 + bx * 128 + u * 2;
        Bls[(u * 2    ) * 264 + k] = src[0];
        Bls[(u * 2 + 1) * 264 + k] = src[1];
    }
    __syncthreads();

    const int lane = tid & 31, wv = tid >> 5;
    const int wm = wv >> 2, wn = wv & 3;        // wave: rows [wm*64,+64), cols [wn*32,+32)
    const int m = lane & 15, half = lane >> 4;

    v8f acc[4][2];
#pragma unroll
    for (int mt = 0; mt < 4; ++mt)
#pragma unroll
        for (int nt = 0; nt < 2; ++nt) acc[mt][nt] = v8f_zero();

#pragma unroll 1
    for (int kc = 0; kc < 8; ++kc) {
        v16bf afr[4], bfr[2];
#pragma unroll
        for (int mt = 0; mt < 4; ++mt)
            afr[mt] = ld_frag16(&Als[(wm * 64 + mt * 16 + m) * 264 + kc * 32 + half * 8]);
#pragma unroll
        for (int nt = 0; nt < 2; ++nt)
            bfr[nt] = ld_frag16(&Bls[(wn * 32 + nt * 16 + m) * 264 + kc * 32 + half * 8]);
#pragma unroll
        for (int mt = 0; mt < 4; ++mt)
#pragma unroll
            for (int nt = 0; nt < 2; ++nt)
                acc[mt][nt] = __builtin_amdgcn_wmma_f32_16x16x32_bf16(
                    false, afr[mt], false, bfr[nt], (short)0, acc[mt][nt], false, false);
    }

    // ---- store with input bias ----
#pragma unroll
    for (int mt = 0; mt < 4; ++mt) {
#pragma unroll
        for (int nt = 0; nt < 2; ++nt) {
            int n   = bx * 128 + wn * 32 + nt * 16 + m;   // C/D col = lane&15
            int dir = n >= H3;
            int nn  = n - dir * H3;
            float bias = dir ? bb0[nn] : bf0[nn];
#pragma unroll
            for (int v = 0; v < 8; ++v) {
                int grow = by * 128 + wm * 64 + mt * 16 + v + 8 * half; // C/D row
                xp[(size_t)grow * H6 + n] = acc[mt][nt][v] + bias;
            }
        }
    }
}

// ---------------------------------------------------------------------------
// Kernel 3: persistent masked GRU scan, one workgroup per (direction, 16-batch
// chunk). h kept in LDS (bf16 for WMMA A, fp32 master). U streamed from L2
// every step (bf16, N-major) straight into WMMA B fragments. The per-step
// gate-input tile xp[t, b0:b0+16, dir*1536 : +1536] (16 rows x 1536 f32,
// row stride 3072 f32) is DMA'd into LDS by the Tensor Data Mover, issued at
// the top of the step so it overlaps the A-fragment loads and the first
// h-tile's WMMAs; s_wait_tensorcnt + barrier fence it before first use.
// ---------------------------------------------------------------------------
__global__ __launch_bounds__(256) void scan_kernel(
    const float* __restrict__ xp, const float* __restrict__ bfv,
    const float* __restrict__ bbv, const __bf16* __restrict__ Ut,
    const int* __restrict__ tokens, float* __restrict__ enc_out,
    float* __restrict__ ysb, float* __restrict__ hf, float* __restrict__ hb) {
    __shared__ __bf16 hbf[16 * 520];   // h as bf16, padded stride
    __shared__ float  h32[16 * 520];   // h master copy (fp32)
    __shared__ float  xpl[16 * H3];    // TDM-staged xp tile (96 KB)

    const int tid = threadIdx.x;
    const int dir = blockIdx.x >> 2;            // 0 = forward, 1 = backward
    const int b0  = (blockIdx.x & 3) * 16;      // batch chunk base

    for (int i = tid; i < 16 * 520; i += 256) { hbf[i] = (__bf16)0.0f; h32[i] = 0.0f; }
    __syncthreads();

    const int lane = tid & 31, wv = tid >> 5;
    const int m = lane & 15, half = lane >> 4;
    const int jb = wv * 64;                          // this wave's h-column base
    const float* brec   = (dir ? bbv : bfv) + H3;    // recurrent bias row
    const __bf16* Ubase = Ut + (size_t)dir * H3 * H_;
    float* ydst = dir ? ysb : enc_out;
    const unsigned lds_xpl = (unsigned)(size_t)(&xpl[0]);   // LDS byte offset

#pragma unroll 1
    for (int step = 0; step < T_; ++step) {
        const int t = dir ? (T_ - 1 - step) : step;

        // ---- issue TDM load of this step's xp tile (wave 0 only) ----
        if (wv == 0) {
            unsigned long long ga =
                (unsigned long long)(size_t)(xp + ((size_t)t * B_ + b0) * H6 + (size_t)dir * H3);
            u32x4 g0;
            g0[0] = 1u;                                   // count=1, user descriptor
            g0[1] = lds_xpl;                              // lds_addr
            g0[2] = (unsigned)(ga & 0xFFFFFFFFu);         // global_addr[31:0]
            g0[3] = (unsigned)((ga >> 32) & 0x1FFFFFFu)   // global_addr[56:32]
                  | (2u << 30);                           // type=2 (image)
            i32x8 g1;
            g1[0] = (int)(2u << 16);        // data_size=4B, mask=0, no flags
            g1[1] = (int)(1536u << 16);     // tensor_dim0[15:0]=1536
            g1[2] = (int)(16u << 16);       // tensor_dim0 hi=0 | tensor_dim1[15:0]=16
            g1[3] = (int)(1536u << 16);     // tensor_dim1 hi=0 | tile_dim0=1536
            g1[4] = 16;                     // tile_dim1=16, tile_dim2=0
            g1[5] = H6;                     // tensor_dim0_stride = 3072 (f32 units)
            g1[6] = 0;                      // stride hi | tensor_dim1_stride lo
            g1[7] = 0;
            i32x4 z4;
            z4[0] = 0; z4[1] = 0; z4[2] = 0; z4[3] = 0;   // groups 2/3: unused (2D)
            i32x8 z8;
#pragma unroll
            for (int q = 0; q < 8; ++q) z8[q] = 0;
            __builtin_amdgcn_tensor_load_to_lds(g0, g1, z4, z4, z8, 0);
        }

        // ---- A fragments: all 16 K-chunks of h (read by every wave) ----
        v16bf afr[16];
#pragma unroll
        for (int kc = 0; kc < 16; ++kc)
            afr[kc] = ld_frag16(&hbf[m * 520 + kc * 32 + half * 8]);
        __syncthreads();   // all hbf reads done before anyone writes h_new

#pragma unroll 1
        for (int ht = 0; ht < 4; ++ht) {
            const int ncol = jb + ht * 16;           // h-column tile base
            v8f az = v8f_zero(), ar = v8f_zero(), ah = v8f_zero();
#pragma unroll 1
            for (int kc = 0; kc < 16; ++kc) {
                v16bf bz = ld_frag16(Ubase + (size_t)(          ncol + m) * H_ + kc * 32 + half * 8);
                v16bf br = ld_frag16(Ubase + (size_t)(H_     + ncol + m) * H_ + kc * 32 + half * 8);
                v16bf bh = ld_frag16(Ubase + (size_t)(2 * H_ + ncol + m) * H_ + kc * 32 + half * 8);
                az = __builtin_amdgcn_wmma_f32_16x16x32_bf16(false, afr[kc], false, bz, (short)0, az, false, false);
                ar = __builtin_amdgcn_wmma_f32_16x16x32_bf16(false, afr[kc], false, br, (short)0, ar, false, false);
                ah = __builtin_amdgcn_wmma_f32_16x16x32_bf16(false, afr[kc], false, bh, (short)0, ah, false, false);
            }
            if (ht == 0) {
                // fence the TDM transfer before the first LDS xp read
                if (wv == 0) __builtin_amdgcn_s_wait_tensorcnt(0);
                __syncthreads();
            }
            const int j = ncol + m;                  // this lane's h column
            const float bz_ = brec[j], br_ = brec[H_ + j], bh_ = brec[2 * H_ + j];
#pragma unroll
            for (int v = 0; v < 8; ++v) {
                const int ml = v + 8 * half;         // local batch row 0..15
                const int b  = b0 + ml;
                const float xz  = xpl[ml * H3 + j];
                const float xrv = xpl[ml * H3 + H_ + j];
                const float xh  = xpl[ml * H3 + 2 * H_ + j];
                const float z  = sigmoidf_(xz + az[v] + bz_);
                const float r  = sigmoidf_(xrv + ar[v] + br_);
                const float hh = tanhf_(xh + r * (ah[v] + bh_));
                const float hold = h32[ml * 520 + j];
                float hn = z * hold + (1.0f - z) * hh;
                if (tokens[b * T_ + t] == 0) hn = hold;   // mask: carry state
                h32[ml * 520 + j] = hn;
                hbf[ml * 520 + j] = (__bf16)hn;
                ydst[((size_t)b * T_ + t) * H_ + j] = hn;
            }
        }
        __syncthreads();   // h_new + xpl reads complete before next step's TDM/A-loads
    }

    float* hout = dir ? hb : hf;
    for (int i = tid; i < 16 * H_; i += 256) {
        int ml = i >> 9, j = i & (H_ - 1);
        hout[(b0 + ml) * H_ + j] = h32[ml * 520 + j];
    }
}

// ---------------------------------------------------------------------------
// Kernel 4: enc_seq = ys_f + ys_b ; state = hf + hb
// ---------------------------------------------------------------------------
__global__ void finish_kernel(float* __restrict__ out, const float* __restrict__ ysb,
                              const float* __restrict__ hf, const float* __restrict__ hb) {
    const size_t total = BTH + (size_t)B_ * H_;
    for (size_t i = (size_t)blockIdx.x * blockDim.x + threadIdx.x;
         i < total; i += (size_t)gridDim.x * blockDim.x) {
        if (i < BTH) {
            out[i] += ysb[i];
        } else {
            size_t r = i - BTH;
            out[i] = hf[r] + hb[r];
        }
    }
}

// ---------------------------------------------------------------------------
// launcher
// ---------------------------------------------------------------------------
extern "C" void kernel_launch(void* const* d_in, const int* in_sizes, int n_in,
                              void* d_out, int out_size, void* d_ws, size_t ws_size,
                              hipStream_t stream) {
    const float* emb = (const float*)d_in[0];
    const float* Wf  = (const float*)d_in[1];
    const float* Uf  = (const float*)d_in[2];
    const float* bfv = (const float*)d_in[3];
    const float* Wb  = (const float*)d_in[4];
    const float* Ub  = (const float*)d_in[5];
    const float* bbv = (const float*)d_in[6];
    const int*   tok = (const int*)d_in[7];

    char* ws = (char*)d_ws;
    size_t off = 0;
    float* xp  = (float*)(ws + off); off += (size_t)BT * H6 * 4;     // 402.6 MB
    float* ysb = (float*)(ws + off); off += BTH * 4;                 //  67.1 MB
    float* hf  = (float*)(ws + off); off += (size_t)B_ * H_ * 4;
    float* hb  = (float*)(ws + off); off += (size_t)B_ * H_ * 4;
    __bf16* Wcat = (__bf16*)(ws + off); off += (size_t)E_ * H6 * 2;
    __bf16* Ut   = (__bf16*)(ws + off); off += (size_t)2 * H3 * H_ * 2;

    float* out = (float*)d_out;

    prep_kernel<<<dim3(2048), dim3(256), 0, stream>>>(Wf, Wb, Uf, Ub, Wcat, Ut);
    gemm1_kernel<<<dim3(24, 256), dim3(256), 0, stream>>>(emb, Wcat, bfv, bbv, tok, xp);
    scan_kernel<<<dim3(8), dim3(256), 0, stream>>>(xp, bfv, bbv, Ut, tok, out, ysb, hf, hb);
    finish_kernel<<<dim3(4096), dim3(256), 0, stream>>>(out, ysb, hf, hb);
}
